// ReliabilityGatedCrossAttention_76605036692072
// MI455X (gfx1250) — matrix operand order repaired and checked by reading, not compiled, not run on hardware
//
#include <hip/hip_runtime.h>

typedef _Float16 half_t;
typedef __attribute__((ext_vector_type(16))) _Float16 v16h;
typedef __attribute__((ext_vector_type(8)))  _Float16 v8h;
typedef __attribute__((ext_vector_type(8)))  float    v8f;

#define DIM   256
#define NH    4
#define HD    64
#define BATCH 4
#define NQ    2048
#define NKV   2048
#define ATT_SCALE 0.125f

// ---------------------------------------------------------------- helpers

__device__ __forceinline__ v8f zero8() {
  v8f z;
#pragma unroll
  for (int e = 0; e < 8; ++e) z[e] = 0.0f;
  return z;
}

__device__ __forceinline__ v8f wmma_f16(v16h a, v16h b, v8f c) {
  // D = A(16x32 f16) * B(32x16 f16) + C(16x16 f32)
  return __builtin_amdgcn_wmma_f32_16x16x32_f16(
      /*neg_a=*/false, a, /*neg_b=*/false, b,
      /*c_mod=*/(short)0, c, /*reuse_a=*/false, /*reuse_b=*/false);
}

// A-fragment (16x32 f16) from a row-major f16 row pointer (points at the
// start of this lane's row within the current 32-wide K chunk).
// Lane l (hi = l>>4): a[e] = row[hi*16 + e]
__device__ __forceinline__ v16h load_a_frag(const half_t* __restrict__ rowp, int hi) {
  v8h l0 = *reinterpret_cast<const v8h*>(rowp + hi * 16);
  v8h l1 = *reinterpret_cast<const v8h*>(rowp + hi * 16 + 8);
  v16h a;
#pragma unroll
  for (int e = 0; e < 8; ++e) { a[e] = l0[e]; a[8 + e] = l1[e]; }
  return a;
}

// ---------------------------------------------------------------- kernel 0a
// Transpose+convert four 256x256 f32 weight matrices to f16, stored as
// WT[out][in] so that WMMA B-fragments are contiguous 32B loads.
__global__ __launch_bounds__(256) void rga_wcvt(
    const float* __restrict__ Wq, const float* __restrict__ Wk,
    const float* __restrict__ Wv, const float* __restrict__ Wo,
    half_t* __restrict__ WqT, half_t* __restrict__ WkT,
    half_t* __restrict__ WvT, half_t* __restrict__ WoT) {
  int tid = blockIdx.x * blockDim.x + threadIdx.x;   // 0 .. 4*65536-1
  int m   = tid >> 16;
  int idx = tid & 65535;
  int o = idx >> 8, i = idx & 255;
  const float* src = (m == 0) ? Wq : (m == 1) ? Wk : (m == 2) ? Wv : Wo;
  half_t*      dst = (m == 0) ? WqT : (m == 1) ? WkT : (m == 2) ? WvT : WoT;
  dst[idx] = (half_t)src[i * DIM + o];
}

// ---------------------------------------------------------------- kernel 0b
__global__ __launch_bounds__(256) void rga_logr(
    const float* __restrict__ rel, float* __restrict__ logr) {
  int tid = blockIdx.x * blockDim.x + threadIdx.x;   // 0 .. B*NKV-1
  logr[tid] = __logf(fmaxf(rel[tid], 1e-6f));
}

// ---------------------------------------------------------------- kernel 1
// Projection GEMM: Y = X(8192x256 f32) @ W(256x256) + bias, optional
// per-row reliability scale, output as f16 in head-major layout:
//   vtrans==0 : dst[((b*NH+h)*N + n)*HD + d]      (q, k)
//   vtrans==1 : dst[((b*NH+h)*HD + d)*NKV + n]    (v transposed)
__global__ __launch_bounds__(128) void rga_proj(
    const float* __restrict__ X, const half_t* __restrict__ WT,
    const float* __restrict__ bias, const float* __restrict__ rscale,
    half_t* __restrict__ dst, int vtrans) {
  int lane = threadIdx.x & 31, wv = threadIdx.x >> 5;
  int lo = lane & 15, hi = lane >> 4;
  int rowBase = blockIdx.x * 16;
  int colBase = (blockIdx.y * 4 + wv) * 16;

  const float*  xrow = X  + (size_t)(rowBase + lo) * DIM;
  const half_t* wrow = WT + (size_t)(colBase + lo) * DIM;

  v8f acc = zero8();
#pragma unroll
  for (int kk = 0; kk < 8; ++kk) {
    int k0 = kk * 32;
    v16h a;
#pragma unroll
    for (int e = 0; e < 8; ++e) {
      a[e]     = (half_t)xrow[k0 + hi * 16 + e];
      a[8 + e] = (half_t)xrow[k0 + hi * 16 + 8 + e];
    }
    v16h b = *reinterpret_cast<const v16h*>(wrow + k0 + hi * 16);
    acc = wmma_f16(a, b, acc);
  }

  float bcol = bias[colBase + lo];
  int col = colBase + lo;
  int h = col >> 6, d = col & 63;
#pragma unroll
  for (int i = 0; i < 8; ++i) {
    int r = rowBase + hi * 8 + i;            // flat row over [B][N]
    float v = acc[i] + bcol;
    if (rscale) v *= rscale[r];
    int bidx = r >> 11, n = r & (NQ - 1);
    if (!vtrans)
      dst[((size_t)(bidx * NH + h) * NQ + n) * HD + d] = (half_t)v;
    else
      dst[((size_t)(bidx * NH + h) * HD + d) * NKV + n] = (half_t)v;
  }
}

// ---------------------------------------------------------------- kernel 2
// Fused attention: softmax(q k^T * s + log r) -> out heads (f16) + attn mean.
// One wave owns a 16-row q tile; 4 waves per block; grid = (NQ/64, B).
__global__ __launch_bounds__(128) void rga_attn(
    const half_t* __restrict__ qh, const half_t* __restrict__ kh,
    const half_t* __restrict__ vhT, const float* __restrict__ logr,
    float* __restrict__ amean, half_t* __restrict__ ohh) {
  int lane = threadIdx.x & 31, wv = threadIdx.x >> 5;
  int lo = lane & 15, hi = lane >> 4;
  int b = blockIdx.y;
  int qbase = blockIdx.x * 64 + wv * 16;

  __shared__ half_t pls[4][16 * 40];          // per-wave 16x32 P tile (pad 40)
  half_t* myp = pls[wv];

  float mrow[NH][8], rs[NH][8];

  // ---- Pass 0: running row-max and sum-exp across all kv ----
#pragma unroll
  for (int h = 0; h < NH; ++h) {
    const half_t* qrow = qh + ((size_t)(b * NH + h) * NQ + qbase + lo) * HD;
    v16h qa0 = load_a_frag(qrow, hi);
    v16h qa1 = load_a_frag(qrow + 32, hi);
    float m8[8], s8[8];
#pragma unroll
    for (int i = 0; i < 8; ++i) { m8[i] = -1e30f; s8[i] = 0.0f; }

    for (int jt = 0; jt < NKV / 16; ++jt) {
      const half_t* krow = kh + ((size_t)(b * NH + h) * NKV + jt * 16 + lo) * HD;
      v16h kb0 = *reinterpret_cast<const v16h*>(krow + hi * 16);
      v16h kb1 = *reinterpret_cast<const v16h*>(krow + 32 + hi * 16);
      v8f c = zero8();
      c = wmma_f16(qa0, kb0, c);
      c = wmma_f16(qa1, kb1, c);
      float lr = logr[b * NKV + jt * 16 + lo];
#pragma unroll
      for (int i = 0; i < 8; ++i) {
        float x = c[i] * ATT_SCALE + lr;
        float tm = x;
#pragma unroll
        for (int off = 1; off < 16; off <<= 1)
          tm = fmaxf(tm, __shfl_xor(tm, off, 16));
        float nm = fmaxf(m8[i], tm);
        float e = __expf(x - nm);
#pragma unroll
        for (int off = 1; off < 16; off <<= 1)
          e += __shfl_xor(e, off, 16);
        s8[i] = s8[i] * __expf(m8[i] - nm) + e;
        m8[i] = nm;
      }
    }
#pragma unroll
    for (int i = 0; i < 8; ++i) { mrow[h][i] = m8[i]; rs[h][i] = 1.0f / s8[i]; }
  }

  // ---- Pass 1: probabilities, P@V, attn mean ----
  v8f oacc[NH][4];
#pragma unroll
  for (int h = 0; h < NH; ++h)
#pragma unroll
    for (int dt = 0; dt < 4; ++dt) oacc[h][dt] = zero8();

  for (int jc = 0; jc < NKV / 32; ++jc) {
    int j0 = jc * 32;
    if (j0 + 32 < NKV)   // emits global_prefetch_b8 for the next K chunk
      __builtin_prefetch(kh + ((size_t)(b * NH) * NKV + j0 + 32 + lane) * HD, 0, 1);

    float pm[2][8];
#pragma unroll
    for (int t = 0; t < 2; ++t)
#pragma unroll
      for (int i = 0; i < 8; ++i) pm[t][i] = 0.0f;

#pragma unroll
    for (int h = 0; h < NH; ++h) {
      const half_t* qrow = qh + ((size_t)(b * NH + h) * NQ + qbase + lo) * HD;
      v16h qa0 = load_a_frag(qrow, hi);
      v16h qa1 = load_a_frag(qrow + 32, hi);

#pragma unroll
      for (int t = 0; t < 2; ++t) {
        int jb = j0 + t * 16;
        const half_t* krow = kh + ((size_t)(b * NH + h) * NKV + jb + lo) * HD;
        v16h kb0 = *reinterpret_cast<const v16h*>(krow + hi * 16);
        v16h kb1 = *reinterpret_cast<const v16h*>(krow + 32 + hi * 16);
        v8f c = zero8();
        c = wmma_f16(qa0, kb0, c);
        c = wmma_f16(qa1, kb1, c);
        float lr = logr[b * NKV + jb + lo];
#pragma unroll
        for (int i = 0; i < 8; ++i) {
          float p = __expf(c[i] * ATT_SCALE + lr - mrow[h][i]) * rs[h][i];
          pm[t][i] += p;
          // C-layout element (row hi*8+i, col t*16+lo) -> LDS 16x32 tile
          myp[(hi * 8 + i) * 40 + t * 16 + lo] = (half_t)p;
        }
      }
      // Re-shape LDS P tile into an A fragment (same wave: DS ops in-order)
      v16h pa;
#pragma unroll
      for (int e = 0; e < 8; ++e) {
        pa[e]     = myp[lo * 40 + hi * 16 + e];
        pa[8 + e] = myp[lo * 40 + hi * 16 + 8 + e];
      }
#pragma unroll
      for (int dt = 0; dt < 4; ++dt) {
        const half_t* vrow =
            vhT + ((size_t)(b * NH + h) * HD + dt * 16 + lo) * NKV + j0 + hi * 16;
        v16h vb = *reinterpret_cast<const v16h*>(vrow);
        oacc[h][dt] = wmma_f16(pa, vb, oacc[h][dt]);
      }
    }

    // attn_mean tile (mean over the 4 heads), written exactly once
#pragma unroll
    for (int t = 0; t < 2; ++t)
#pragma unroll
      for (int i = 0; i < 8; ++i) {
        size_t row = (size_t)b * NQ + qbase + hi * 8 + i;
        amean[row * NKV + j0 + t * 16 + lo] = pm[t][i] * 0.25f;
      }
  }

  // head outputs -> [b][q][256] f16 (ready for output projection)
#pragma unroll
  for (int h = 0; h < NH; ++h)
#pragma unroll
    for (int dt = 0; dt < 4; ++dt)
#pragma unroll
      for (int i = 0; i < 8; ++i) {
        int row = qbase + hi * 8 + i;
        int col = h * HD + dt * 16 + lo;
        ohh[((size_t)b * NQ + row) * DIM + col] = (half_t)oacc[h][dt][i];
      }
}

// ---------------------------------------------------------------- kernel 3
// Output projection: out = OHH(8192x256 f16) @ Wo + bo  (f32 out)
__global__ __launch_bounds__(128) void rga_outproj(
    const half_t* __restrict__ A, const half_t* __restrict__ WT,
    const float* __restrict__ bias, float* __restrict__ out) {
  int lane = threadIdx.x & 31, wv = threadIdx.x >> 5;
  int lo = lane & 15, hi = lane >> 4;
  int rowBase = blockIdx.x * 16;
  int colBase = (blockIdx.y * 4 + wv) * 16;

  const half_t* arow = A  + (size_t)(rowBase + lo) * DIM;
  const half_t* wrow = WT + (size_t)(colBase + lo) * DIM;

  v8f acc = zero8();
#pragma unroll
  for (int kk = 0; kk < 8; ++kk) {
    v16h a = load_a_frag(arow + kk * 32, hi);
    v16h b = *reinterpret_cast<const v16h*>(wrow + kk * 32 + hi * 16);
    acc = wmma_f16(a, b, acc);
  }
  float bcol = bias[colBase + lo];
#pragma unroll
  for (int i = 0; i < 8; ++i)
    out[((size_t)(rowBase + hi * 8 + i)) * DIM + colBase + lo] = acc[i] + bcol;
}

// ---------------------------------------------------------------- launch

extern "C" void kernel_launch(void* const* d_in, const int* in_sizes, int n_in,
                              void* d_out, int out_size, void* d_ws, size_t ws_size,
                              hipStream_t stream) {
  (void)in_sizes; (void)n_in; (void)out_size; (void)ws_size;

  const float* qf  = (const float*)d_in[0];   // [B][NQ][DIM]
  const float* kvf = (const float*)d_in[1];   // [B][NKV][DIM]
  const float* rel = (const float*)d_in[2];   // [B][NKV]
  const float* Wq  = (const float*)d_in[3];
  const float* bq  = (const float*)d_in[4];
  const float* Wk  = (const float*)d_in[5];
  const float* bk  = (const float*)d_in[6];
  const float* Wv  = (const float*)d_in[7];
  const float* bv  = (const float*)d_in[8];
  const float* Wo  = (const float*)d_in[9];
  const float* bo  = (const float*)d_in[10];

  char* ws = (char*)d_ws;
  const size_t WSZ = (size_t)DIM * DIM * sizeof(half_t);        // 128 KB
  half_t* WqT  = (half_t*)(ws + 0 * WSZ);
  half_t* WkT  = (half_t*)(ws + 1 * WSZ);
  half_t* WvT  = (half_t*)(ws + 2 * WSZ);
  half_t* WoT  = (half_t*)(ws + 3 * WSZ);
  float*  logr = (float*)(ws + 4 * WSZ);                        // 32 KB
  size_t off = 4 * WSZ + (size_t)BATCH * NKV * sizeof(float);
  const size_t HSZ = (size_t)BATCH * NH * NQ * HD * sizeof(half_t); // 4 MB
  half_t* qh  = (half_t*)(ws + off);           off += HSZ;
  half_t* kh  = (half_t*)(ws + off);           off += HSZ;
  half_t* vhT = (half_t*)(ws + off);           off += HSZ;
  half_t* ohh = (half_t*)(ws + off);           // [B][NQ][DIM] f16, 4 MB

  float* out   = (float*)d_out;                                // [B][NQ][DIM]
  float* amean = out + (size_t)BATCH * NQ * DIM;               // [B][NQ][NKV]

  // 0) weight convert/transpose + log reliability
  rga_wcvt<<<dim3(4 * 65536 / 256), dim3(256), 0, stream>>>(
      Wq, Wk, Wv, Wo, WqT, WkT, WvT, WoT);
  rga_logr<<<dim3(BATCH * NKV / 256), dim3(256), 0, stream>>>(rel, logr);

  // 1) projections (q; k,v gated by reliability; v stored transposed)
  dim3 pg(BATCH * NQ / 16, DIM / 64);
  rga_proj<<<pg, dim3(128), 0, stream>>>(qf,  WqT, bq, nullptr, qh,  0);
  rga_proj<<<pg, dim3(128), 0, stream>>>(kvf, WkT, bk, rel,     kh,  0);
  rga_proj<<<pg, dim3(128), 0, stream>>>(kvf, WvT, bv, rel,     vhT, 1);

  // 2) fused attention + attn_mean
  rga_attn<<<dim3(NQ / 64, BATCH), dim3(128), 0, stream>>>(
      qh, kh, vhT, logr, amean, ohh);

  // 3) output projection
  rga_outproj<<<dim3(BATCH * NQ / 16, DIM / 64), dim3(128), 0, stream>>>(
      ohh, WoT, bo, out);
}
